// CRF_79568564126527
// MI455X (gfx1250) — matrix operand order repaired and checked by reading, not compiled, not run on hardware
//
#include <hip/hip_runtime.h>
#include <math.h>

// CRF forward: sum_b ( joint_llh[b] - log_Z[b] )
// S=512 timesteps, B=256 batch, T=128 tags.
//
// log_Z scan runs in the *log2 domain* and is turned into WMMA f16 GEMMs via
// the exp-space trick:
//   lp2'[b,j] = M2[b] + log2( sum_i 2^(lp2[b,i]-M2[b]) * E[i,j] ) + em[s,b,j]*log2e
// with E = exp(trans) constant over time -> each wave keeps its B tile in VGPRs.

#define S_LEN 512
#define BATCH 256
#define NTAG  128

#define LP_STRIDE 132   // f32 lp row stride (column access hits distinct banks)
#define PH_STRIDE 136   // f16 P row stride (68 dwords -> 4*M % 64 distinct banks)

#define LOG2E 1.4426950408889634f
#define LN2   0.6931471805599453f

typedef __attribute__((ext_vector_type(16))) _Float16 v16h;
typedef __attribute__((ext_vector_type(8)))  float    v8f;

__device__ __forceinline__ float fast_exp2(float x) {
#if defined(__HIP_DEVICE_COMPILE__) && __has_builtin(__builtin_amdgcn_exp2f)
    return __builtin_amdgcn_exp2f(x);        // raw v_exp_f32
#else
    return exp2f(x);
#endif
}
__device__ __forceinline__ float fast_log2(float x) {
#if defined(__HIP_DEVICE_COMPILE__) && __has_builtin(__builtin_amdgcn_logf)
    return __builtin_amdgcn_logf(x);         // raw v_log_f32
#else
    return log2f(x);
#endif
}

// ---------------------------------------------------------------- numerator
// One block per batch element; timesteps split across 256 threads; all terms
// are sums over s, reduced with a fixed-order LDS tree (deterministic).
__global__ __launch_bounds__(256)
void crf_numerator(const float* __restrict__ em, const int* __restrict__ tags,
                   const int* __restrict__ mask, const float* __restrict__ startT,
                   const float* __restrict__ endT, const float* __restrict__ trans,
                   float* __restrict__ llh_out) {
    __shared__ float sbuf[256];
    __shared__ int   mbuf[256];
    const int b = blockIdx.x;
    const int t = threadIdx.x;

    float part = 0.f;
    int   mcnt = 0;
#pragma unroll
    for (int s = t; s < S_LEN; s += 256) {
        int   tg = tags[s * BATCH + b];
        int   mk = mask[s * BATCH + b];
        float mf = (float)mk;
        part += em[((size_t)s * BATCH + b) * NTAG + tg] * mf;        // emissions
        if (s > 0)
            part += trans[tags[(s - 1) * BATCH + b] * NTAG + tg] * mf; // transitions
        mcnt += mk;
    }
    sbuf[t] = part;
    mbuf[t] = mcnt;
    __syncthreads();
    for (int off = 128; off > 0; off >>= 1) {
        if (t < off) { sbuf[t] += sbuf[t + off]; mbuf[t] += mbuf[t + off]; }
        __syncthreads();
    }
    if (t == 0) {
        int last_idx = mbuf[0] - 1;           // mask[0]==1 guaranteed by setup
        llh_out[b] = sbuf[0] + startT[tags[b]] + endT[tags[last_idx * BATCH + b]];
    }
}

// ------------------------------------------------------------- log partition
__global__ __launch_bounds__(256, 1)
void crf_partition(const float* __restrict__ em, const int* __restrict__ mask,
                   const float* __restrict__ startT, const float* __restrict__ endT,
                   const float* __restrict__ trans, float* __restrict__ logz_out) {
    __shared__ float    lpS[16 * LP_STRIDE];   // lp[b_tile][tag], log2 units, fp32
    __shared__ _Float16 pH [16 * PH_STRIDE];   // 2^(lp - rowmax), f16
    __shared__ float    rowm[16];

    const int tid  = threadIdx.x;
    const int lane = tid & 31;
    const int wave = tid >> 5;                 // 0..7 -> 16-col tile of next tags
    const int b0   = blockIdx.x * 16;          // 16 batch rows per block

    // ---- constant B operands: E[k][n] = exp(trans[k][n]) for this wave's tile.
    // B k order used here (contiguous, same as A): lane%16 = N, lanes>=16 hold K+16,
    // half h (0..15) -> K = h within the half-wave's 16-row K slab.
    const int ncol = wave * 16 + (lane & 15);
    const int khiB = (lane >= 16) ? 16 : 0;
    v16h Bop[4];
#pragma unroll
    for (int kc = 0; kc < 4; ++kc) {
#pragma unroll
        for (int h = 0; h < 16; ++h) {
            int k = kc * 32 + khiB + h;
            Bop[kc][h] = (_Float16)fast_exp2(trans[k * NTAG + ncol] * LOG2E);
        }
    }

    const int r16 = tid >> 4;     // row (0..15) owned by this 16-lane half-wave
    const int seg = tid & 15;     // 8-wide column segment within that row
    const int rhi = (lane >= 16) ? 8 : 0;

    // ---- lp[0] = (startT + emissions[0]) * log2e
#pragma unroll
    for (int i = 0; i < 8; ++i) {
        int c = seg * 8 + i;
        lpS[r16 * LP_STRIDE + c] =
            (startT[c] + em[((size_t)(b0 + r16)) * NTAG + c]) * LOG2E;
    }
    __syncthreads();

    // each lane owns lp entries (row r+rhi, col ncol); cache them in registers
    float lpcol[8];
#pragma unroll
    for (int r = 0; r < 8; ++r)
        lpcol[r] = lpS[(r + rhi) * LP_STRIDE + ncol];

#pragma unroll 1
    for (int s = 1; s < S_LEN; ++s) {
        // issue step-s loads early; consumed after the WMMAs
        float emv[8];
#pragma unroll
        for (int r = 0; r < 8; ++r)
            emv[r] = em[((size_t)s * BATCH + b0 + r + rhi) * NTAG + ncol];
        const int4 mk0 = *(const int4*)&mask[s * BATCH + b0 + rhi];
        const int4 mk1 = *(const int4*)&mask[s * BATCH + b0 + rhi + 4];
        float mskf[8];
        mskf[0] = (mk0.x > 0) ? 1.0f : 0.0f;
        mskf[1] = (mk0.y > 0) ? 1.0f : 0.0f;
        mskf[2] = (mk0.z > 0) ? 1.0f : 0.0f;
        mskf[3] = (mk0.w > 0) ? 1.0f : 0.0f;
        mskf[4] = (mk1.x > 0) ? 1.0f : 0.0f;
        mskf[5] = (mk1.y > 0) ? 1.0f : 0.0f;
        mskf[6] = (mk1.z > 0) ? 1.0f : 0.0f;
        mskf[7] = (mk1.w > 0) ? 1.0f : 0.0f;

        // 1) row r16: load once, max via 8 local + 4 shfl_xor (all 16 lanes of
        // the half-wave end with the row max in-register), then P = 2^(lp-mx).
        float lpv[8];
#pragma unroll
        for (int i = 0; i < 8; ++i)
            lpv[i] = lpS[r16 * LP_STRIDE + seg * 8 + i];
        float mx = lpv[0];
#pragma unroll
        for (int i = 1; i < 8; ++i) mx = fmaxf(mx, lpv[i]);
#pragma unroll
        for (int off = 1; off < 16; off <<= 1)
            mx = fmaxf(mx, __shfl_xor(mx, off, 32));
        if ((lane & 15) == 0) rowm[r16] = mx;    // for step 3 (other rows)
#pragma unroll
        for (int i = 0; i < 8; ++i)
            pH[r16 * PH_STRIDE + seg * 8 + i] = (_Float16)fast_exp2(lpv[i] - mx);
        __syncthreads();   // barrier A: pH + rowm published

        // 2) acc(16x16) = P(16x128) x E(128x16) via 4 chained WMMAs.
        // A k order used here (contiguous, same as B): lane%16 = M; lanes<16 hold K {0..15},
        // lanes>=16 hold K {16..31}.
        const int mrow = lane & 15;
        const int khiA = (lane >= 16) ? 16 : 0;
        v8f acc = {};
#pragma unroll
        for (int kc = 0; kc < 4; ++kc) {
            v16h a;
#pragma unroll
            for (int h = 0; h < 16; ++h) {
                int k = kc * 32 + khiA + h;
                a[h] = pH[mrow * PH_STRIDE + k];
            }
            acc = __builtin_amdgcn_wmma_f32_16x16x32_f16(
                false, a, false, Bop[kc], (short)0, acc, false, false);
        }

        // 3) mask-gated update as pure arithmetic (lerp with 0/1 factor) so the
        // compiler cannot sink the log under a divergent branch.
        // rowm broadcast: two float4 LDS loads, same address across lanes.
        const float4 rm0 = *(const float4*)&rowm[rhi];
        const float4 rm1 = *(const float4*)&rowm[rhi + 4];
        const float rmv[8] = {rm0.x, rm0.y, rm0.z, rm0.w,
                              rm1.x, rm1.y, rm1.z, rm1.w};
#pragma unroll
        for (int r = 0; r < 8; ++r) {
            float nv = rmv[r] + fast_log2(acc[r]) + emv[r] * LOG2E;
            lpcol[r] += mskf[r] * (nv - lpcol[r]);   // mask ? nv : old
            lpS[(r + rhi) * LP_STRIDE + ncol] = lpcol[r];
        }

        // prefetch next step's emission tile (gfx1250 global_prefetch_b8)
        {
            int sn = (s + 1 < S_LEN) ? (s + 1) : (S_LEN - 1);
            if (lane == 0)
                __builtin_prefetch(&em[((size_t)sn * BATCH + b0 + wave * 2) * NTAG], 0, 1);
        }
        __syncthreads();   // barrier B: lp published for next step
    }

    // ---- log_z[b] = ln2 * log2sumexp2_t( lp2[b,t] + endT[t]*log2e )
    // (half-wave shfl reductions only; no LDS needed)
    float ev[8];
#pragma unroll
    for (int i = 0; i < 8; ++i) {
        int c = seg * 8 + i;
        ev[i] = lpS[r16 * LP_STRIDE + c] + endT[c] * LOG2E;
    }
    float mx = ev[0];
#pragma unroll
    for (int i = 1; i < 8; ++i) mx = fmaxf(mx, ev[i]);
#pragma unroll
    for (int off = 1; off < 16; off <<= 1)
        mx = fmaxf(mx, __shfl_xor(mx, off, 32));
    float sum = 0.f;
#pragma unroll
    for (int i = 0; i < 8; ++i) sum += fast_exp2(ev[i] - mx);
#pragma unroll
    for (int off = 1; off < 16; off <<= 1)
        sum += __shfl_xor(sum, off, 32);
    if ((lane & 15) == 0)
        logz_out[b0 + r16] = (mx + fast_log2(sum)) * LN2;
}

// ------------------------------------------------- deterministic final reduce
__global__ __launch_bounds__(256)
void crf_reduce(const float* __restrict__ llh, const float* __restrict__ logz,
                float* __restrict__ out) {
    __shared__ float buf[256];
    int t = threadIdx.x;
    buf[t] = llh[t] - logz[t];
    __syncthreads();
    for (int off = 128; off > 0; off >>= 1) {
        if (t < off) buf[t] += buf[t + off];
        __syncthreads();
    }
    if (t == 0) out[0] = buf[0];
}

// ---------------------------------------------------------------------- launch
extern "C" void kernel_launch(void* const* d_in, const int* in_sizes, int n_in,
                              void* d_out, int out_size, void* d_ws, size_t ws_size,
                              hipStream_t stream) {
    const float* emissions = (const float*)d_in[0];   // (S,B,T)
    const int*   tags      = (const int*)  d_in[1];   // (S,B)
    const int*   mask      = (const int*)  d_in[2];   // (S,B)
    const float* startT    = (const float*)d_in[3];   // (T,)
    const float* endT      = (const float*)d_in[4];   // (T,)
    const float* trans     = (const float*)d_in[5];   // (T,T)

    float* llh  = (float*)d_ws;                       // [256]
    float* logz = llh + BATCH;                        // [256]
    float* out  = (float*)d_out;

    crf_numerator<<<BATCH, 256, 0, stream>>>(emissions, tags, mask, startT, endT, trans, llh);
    crf_partition<<<BATCH / 16, 256, 0, stream>>>(emissions, mask, startT, endT, trans, logz);
    crf_reduce<<<1, 256, 0, stream>>>(llh, logz, out);
}
